// CLIPSampleHead_29025388987098
// MI455X (gfx1250) — matrix operand order repaired and checked by reading, not compiled, not run on hardware
//
#include <hip/hip_runtime.h>
#include <hip/hip_bf16.h>
#include <math.h>
#include <stdint.h>

typedef _Float16 f16;
typedef _Float16 v16h_t __attribute__((ext_vector_type(16)));
typedef _Float16 v8h_t  __attribute__((ext_vector_type(8)));
typedef float    v8f_t  __attribute__((ext_vector_type(8)));
typedef int      v4i_t  __attribute__((ext_vector_type(4)));

#define DIMC  1024
#define NIMG  384      // B*NB*NC = 2*32*6
#define CROPSZ 224
#define IMGH  512
#define IMGW  1408

// Async global->LDS path (gfx1250): gated so the file still compiles (with the
// register-staging fallback) if the toolchain does not declare the builtins.
// Probe round revealed the signature: (global v4i* src, lds v4i* dst, Ii, Ii).
#if defined(__HIP_DEVICE_COMPILE__) && __has_builtin(__builtin_amdgcn_global_load_async_to_lds_b128)
#define USE_ASYNC 1
#define ASG __attribute__((address_space(1)))
#define ASL __attribute__((address_space(3)))
__device__ __forceinline__ void wait_async0() {
#if __has_builtin(__builtin_amdgcn_s_wait_asynccnt)
  __builtin_amdgcn_s_wait_asynccnt(0);
#else
  asm volatile("s_wait_asynccnt 0x0" ::: "memory");
#endif
}
#else
#define USE_ASYNC 0
#endif

// ---------------------------------------------------------------------------
// Setup: projection -> theta (2x3 affine per (box,cam)) + valid mask
// ---------------------------------------------------------------------------
__global__ void setup_kernel(const float* __restrict__ boxes, const float* __restrict__ la,
                             const float* __restrict__ l2i, const float* __restrict__ aug,
                             const float* __restrict__ tw, const float* __restrict__ tb,
                             float* __restrict__ theta, int* __restrict__ valid) {
  int idx = blockIdx.x * 64 + threadIdx.x;
  if (idx >= NIMG) return;
  int n = idx / 6, c = idx - n * 6;
  int b = n >> 5;                                  // NB = 32
  const float* bx = boxes + (size_t)n * 9;
  const float* L  = la + (size_t)b * 16;
  float p0 = bx[0] - L[3], p1 = bx[1] - L[7], p2 = bx[2] - L[11];
  float a00=L[0],a01=L[1],a02=L[2],a10=L[4],a11=L[5],a12=L[6],a20=L[8],a21=L[9],a22=L[10];
  float det = a00*(a11*a22-a12*a21) - a01*(a10*a22-a12*a20) + a02*(a10*a21-a11*a20);
  float id = 1.f / det;
  float i00=(a11*a22-a12*a21)*id, i01=(a02*a21-a01*a22)*id, i02=(a01*a12-a02*a11)*id;
  float i10=(a12*a20-a10*a22)*id, i11=(a00*a22-a02*a20)*id, i12=(a02*a10-a00*a12)*id;
  float i20=(a10*a21-a11*a20)*id, i21=(a01*a20-a00*a21)*id, i22=(a00*a11-a01*a10)*id;
  float q0=i00*p0+i01*p1+i02*p2, q1=i10*p0+i11*p1+i12*p2, q2=i20*p0+i21*p1+i22*p2;
  const float* M = l2i + (size_t)(b*6 + c) * 16;
  float r0 = M[0]*q0 + M[1]*q1 + M[2]*q2 + M[3];
  float r1 = M[4]*q0 + M[5]*q1 + M[6]*q2 + M[7];
  float r2 = M[8]*q0 + M[9]*q1 + M[10]*q2 + M[11];
  float z = fminf(fmaxf(r2, 1e-5f), 100000.f);
  float u = r0 / z, v = r1 / z;
  const float* A = aug + (size_t)(b*6 + c) * 16;
  float au = A[0]*u + A[1]*v + A[2]*z + A[3];      // x / col coordinate
  float av = A[4]*u + A[5]*v + A[6]*z + A[7];      // y / row coordinate
  valid[idx] = (av < (float)IMGH) && (av >= 0.f) && (au < (float)IMGW) && (au >= 0.f);
  float t[4];
  #pragma unroll
  for (int j = 0; j < 4; ++j) {
    float s = tb[j];
    #pragma unroll
    for (int k = 0; k < 9; ++k) s += bx[k] * tw[j*9 + k];
    t[j] = tanhf(s);
  }
  float ty = av / (float)IMGH * 2.f - 1.f;
  float tx = au / (float)IMGW * 2.f - 1.f;
  float* th = theta + (size_t)idx * 6;
  th[0]=t[0]; th[1]=t[1]; th[2]=tx; th[3]=t[2]; th[4]=t[3]; th[5]=ty;
}

// ---------------------------------------------------------------------------
// Weight conversion f32 -> f16 into padded [N][Kp] (zero fill k>=K).
// ---------------------------------------------------------------------------
__global__ void cvt_pad_kernel(const float* __restrict__ s, f16* __restrict__ d,
                               int N, int K, int Kp) {
  int i = blockIdx.x * 256 + threadIdx.x;
  if (i < N * Kp) {
    int n = i / Kp, k = i - n * Kp;
    d[i] = (k < K) ? (f16)s[(size_t)n * K + k] : (f16)0.f;
  }
}
__global__ void cvtT_pad_kernel(const float* __restrict__ s, f16* __restrict__ d,
                                int N, int K, int Kp) {
  int i = blockIdx.x * 256 + threadIdx.x;
  if (i < N * Kp) {
    int n = i / Kp, k = i - n * Kp;
    d[i] = (k < K) ? (f16)s[(size_t)k * N + n] : (f16)0.f;
  }
}

// ---------------------------------------------------------------------------
// Grid-sample: affine grid + bilinear sample with zero padding -> crops (NHWC f16)
// ---------------------------------------------------------------------------
__device__ __forceinline__ float samp(const float* p, float xf, float yf) {
  bool ok = (xf >= 0.f) && (xf <= (float)(IMGW-1)) && (yf >= 0.f) && (yf <= (float)(IMGH-1));
  int xi = (int)fminf(fmaxf(xf, 0.f), (float)(IMGW-1));
  int yi = (int)fminf(fmaxf(yf, 0.f), (float)(IMGH-1));
  float v = p[(size_t)yi * IMGW + xi];
  return ok ? v : 0.f;
}

__global__ __launch_bounds__(256) void crop_kernel(const float* __restrict__ imgs,
    const float* __restrict__ theta, f16* __restrict__ crops) {
  int img = blockIdx.x;
  int pix = blockIdx.y * 256 + threadIdx.x;
  if (pix >= CROPSZ * CROPSZ) return;
  int y = pix / CROPSZ, x = pix - y * CROPSZ;
  const float* th = theta + (size_t)img * 6;
  float gx = (2.f * x + 1.f) / (float)CROPSZ - 1.f;
  float gy = (2.f * y + 1.f) / (float)CROPSZ - 1.f;
  float sx = th[0]*gx + th[1]*gy + th[2];
  float sy = th[3]*gx + th[4]*gy + th[5];
  float ix = ((sx + 1.f) * (float)IMGW - 1.f) * 0.5f;
  float iy = ((sy + 1.f) * (float)IMGH - 1.f) * 0.5f;
  float x0 = floorf(ix), y0 = floorf(iy);
  float wx = ix - x0, wy = iy - y0;
  int n = img / 6, c = img - n * 6;
  const float* cam = imgs + (size_t)((n >> 5) * 6 + c) * 3 * IMGH * IMGW;
  size_t ob = ((size_t)img * CROPSZ * CROPSZ + pix) * 3;
  #pragma unroll
  for (int ch = 0; ch < 3; ++ch) {
    const float* pp = cam + (size_t)ch * IMGH * IMGW;
    float v00 = samp(pp, x0,     y0);
    float v01 = samp(pp, x0+1.f, y0);
    float v10 = samp(pp, x0,     y0+1.f);
    float v11 = samp(pp, x0+1.f, y0+1.f);
    float r = v00*(1.f-wx)*(1.f-wy) + v01*wx*(1.f-wy) + v10*(1.f-wx)*wy + v11*wx*wy;
    crops[ob + ch] = (f16)r;
  }
}

// ---------------------------------------------------------------------------
// Implicit-GEMM convolution / batched GEMM with WMMA f16 (f32 accumulate).
// act: NHWC f16 [IMG][IH*IW][IC]; wt: [N][Kp] f16 (K = IC*KH*KW, zero-padded);
// out: [IMG][M][N] f16 or f32. 1x1 case == pure row-major GEMM.
// 4 waves / block, 32x32 wave tile (4 wmma per K-step per wave), LDS double
// buffered. With USE_ASYNC, B tiles (and pure-mode A tiles) are copied
// global->LDS by the async engine (ASYNCcnt) while WMMAs run; conv-mode A
// im2col stays on the register path (scatter-gather).
// ---------------------------------------------------------------------------
struct GemmP {
  const f16* act; const f16* wt; void* out;
  int IC, IH, IW, KH, KW, SH, SW, PH, PW, OH, OW, M, N, K, Kp;
  int relu, out_f16;
};

#define BM 64
#define BN 64
#define BK 32
#define AS 40   // LDS row stride in halves (80B) -> conflict-free b128 fragment loads

__global__ __launch_bounds__(128) void gemm_kernel(GemmP p) {
  __shared__ f16 lA[2][BM * AS];
  __shared__ f16 lB[2][BN * AS];
  const int tid  = threadIdx.x;
  const int img  = blockIdx.z;
  const int m0   = blockIdx.x * BM;
  const int n0   = blockIdx.y * BN;
  const int wave = tid >> 5, lane = tid & 31;
  const int half = lane >> 4, r = lane & 15;
  const int wm   = (wave >> 1) * 32;   // wave M offset (0,32)
  const int wn   = (wave & 1) * 32;    // wave N offset (0,32)
  v8f_t c00 = {0.f,0.f,0.f,0.f,0.f,0.f,0.f,0.f};
  v8f_t c01 = c00, c10 = c00, c11 = c00;
  const int KHW = p.KH * p.KW;
  const bool pure = (KHW == 1);
  // staging: thread t handles 16 halves of A-row and B-row (t>>1), cols (t&1)*16+
  const int sr = tid >> 1;
  const int sc = (tid & 1) * 16;
  const int am = m0 + sr;
  const int aoy = am / p.OW;
  const int aox = am - aoy * p.OW;
  const size_t act_img = (size_t)img * p.IH * p.IW;
  const int bn = n0 + sr;              // always < N (N is a multiple of 64 here)
  const f16* wrow_base = p.wt + (size_t)bn * p.Kp;

  alignas(16) f16 areg[16];
  v8h_t breg0, breg1;

  // conv-mode (or out-of-range) A fetch into registers
  auto fetchA = [&](int k0) {
    int kk = k0 + sc;
    if (am < p.M) {
      if (pure) {
        const f16* arow = p.act + (act_img + (size_t)aoy * p.IW + aox) * p.IC + kk;
        *(v8h_t*)&areg[0] = *(const v8h_t*)arow;
        *(v8h_t*)&areg[8] = *(const v8h_t*)(arow + 8);
      } else {
        int ic = kk / KHW;
        int rem = kk - ic * KHW;
        int ky = rem / p.KW;
        int kx = rem - ky * p.KW;
        #pragma unroll
        for (int j = 0; j < 16; ++j) {
          int k = kk + j;
          f16 v = (f16)0.f;
          if (k < p.K) {
            int iy = aoy * p.SH - p.PH + ky;
            int ix = aox * p.SW - p.PW + kx;
            if ((unsigned)iy < (unsigned)p.IH && (unsigned)ix < (unsigned)p.IW)
              v = p.act[(act_img + (size_t)iy * p.IW + ix) * p.IC + ic];
          }
          areg[j] = v;
          ++kx;
          if (kx == p.KW) { kx = 0; ++ky; if (ky == p.KH) { ky = 0; ++ic; } }
        }
      }
    } else {
      #pragma unroll
      for (int j = 0; j < 16; ++j) areg[j] = (f16)0.f;
    }
  };
  auto fetchB = [&](int k0) {
    const f16* wr = wrow_base + k0 + sc;
    breg0 = *(const v8h_t*)wr;
    breg1 = *(const v8h_t*)(wr + 8);
  };
  auto stageA = [&](int bufi) {
    *(v8h_t*)&lA[bufi][sr * AS + sc]     = *(v8h_t*)&areg[0];
    *(v8h_t*)&lA[bufi][sr * AS + sc + 8] = *(v8h_t*)&areg[8];
  };
  auto stageB = [&](int bufi) {
    *(v8h_t*)&lB[bufi][sr * AS + sc]     = breg0;
    *(v8h_t*)&lB[bufi][sr * AS + sc + 8] = breg1;
  };

#if USE_ASYNC
  // async engine copies: B tile always; A tile in pure mode (rows >= M are
  // not fetched async; fetchA/stageA write zeros there instead)
  auto async_fetch = [&](int bufi, int k0) {
    int kk = k0 + sc;
    const f16* wr = wrow_base + kk;
    ASL v4i_t* db = (ASL v4i_t*)&lB[bufi][sr * AS + sc];   // 16B aligned
    __builtin_amdgcn_global_load_async_to_lds_b128(
        (ASG v4i_t*)(uintptr_t)wr, db, 0, 0);
    __builtin_amdgcn_global_load_async_to_lds_b128(
        (ASG v4i_t*)(uintptr_t)(wr + 8), db + 1, 0, 0);
    if (pure && am < p.M) {
      const f16* ar = p.act + (act_img + (size_t)aoy * p.IW + aox) * p.IC + kk;
      ASL v4i_t* da = (ASL v4i_t*)&lA[bufi][sr * AS + sc];
      __builtin_amdgcn_global_load_async_to_lds_b128(
          (ASG v4i_t*)(uintptr_t)ar, da, 0, 0);
      __builtin_amdgcn_global_load_async_to_lds_b128(
          (ASG v4i_t*)(uintptr_t)(ar + 8), da + 1, 0, 0);
    }
  };
#endif

  // ---- prologue: tile 0 ----
#if USE_ASYNC
  async_fetch(0, 0);
  if (!pure || am >= p.M) { fetchA(0); stageA(0); }
  wait_async0();
#else
  fetchA(0); fetchB(0);
  stageA(0); stageB(0);
#endif
  __syncthreads();

  int buf = 0;
  for (int k0 = 0; k0 < p.Kp; k0 += BK) {
    const bool more = (k0 + BK) < p.Kp;
    if (more) {
      if (k0 + 2 * BK < p.Kp)
        __builtin_prefetch(wrow_base + k0 + 2 * BK, 0, 1);   // global_prefetch toward L2
#if USE_ASYNC
      async_fetch(buf ^ 1, k0 + BK);
      if (!pure || am >= p.M) fetchA(k0 + BK);
#else
      fetchA(k0 + BK); fetchB(k0 + BK);
#endif
    }
    // ---- fragments per ISA wave32 layout ----
    const f16* LA = lA[buf];
    const f16* LB = lB[buf];
    v16h_t a0, a1, b0, b1;
    {
      const v8h_t lo0 = *(const v8h_t*)&LA[(wm + r) * AS + half * 8];
      const v8h_t hi0 = *(const v8h_t*)&LA[(wm + r) * AS + half * 8 + 16];
      const v8h_t lo1 = *(const v8h_t*)&LA[(wm + 16 + r) * AS + half * 8];
      const v8h_t hi1 = *(const v8h_t*)&LA[(wm + 16 + r) * AS + half * 8 + 16];
      #pragma unroll
      for (int i = 0; i < 8; ++i) { a0[i]=lo0[i]; a0[i+8]=hi0[i]; a1[i]=lo1[i]; a1[i+8]=hi1[i]; }
    }
    {
      const v8h_t x0 = *(const v8h_t*)&LB[(wn + r) * AS + half * 8];
      const v8h_t x1 = *(const v8h_t*)&LB[(wn + r) * AS + half * 8 + 16];
      const v8h_t y0 = *(const v8h_t*)&LB[(wn + 16 + r) * AS + half * 8];
      const v8h_t y1 = *(const v8h_t*)&LB[(wn + 16 + r) * AS + half * 8 + 16];
      #pragma unroll
      for (int i = 0; i < 8; ++i) { b0[i]=x0[i]; b0[i+8]=x1[i]; b1[i]=y0[i]; b1[i+8]=y1[i]; }
    }
    c00 = __builtin_amdgcn_wmma_f32_16x16x32_f16(false, a0, false, b0, (short)0, c00, false, false);
    c01 = __builtin_amdgcn_wmma_f32_16x16x32_f16(false, a0, false, b1, (short)0, c01, false, false);
    c10 = __builtin_amdgcn_wmma_f32_16x16x32_f16(false, a1, false, b0, (short)0, c10, false, false);
    c11 = __builtin_amdgcn_wmma_f32_16x16x32_f16(false, a1, false, b1, (short)0, c11, false, false);
    if (more) {
#if USE_ASYNC
      if (!pure || am >= p.M) stageA(buf ^ 1);
      wait_async0();
#else
      stageA(buf ^ 1); stageB(buf ^ 1);
#endif
      buf ^= 1;
    }
    __syncthreads();
  }

  // ---- epilogue: D[m][n], m = i + 8*half (+16 for second row group) ----
  #pragma unroll
  for (int i = 0; i < 8; ++i) {
    int mA = m0 + wm + i + 8 * half;
    int mB = mA + 16;
    int na = n0 + wn + r;
    int nb = na + 16;
    float v00 = c00[i], v01 = c01[i], v10 = c10[i], v11 = c11[i];
    if (p.relu) {
      v00 = fmaxf(v00, 0.f); v01 = fmaxf(v01, 0.f);
      v10 = fmaxf(v10, 0.f); v11 = fmaxf(v11, 0.f);
    }
    if (p.out_f16) {
      f16* o = (f16*)p.out;
      if (mA < p.M) {
        size_t rb = ((size_t)img * p.M + mA) * p.N;
        o[rb + na] = (f16)v00; o[rb + nb] = (f16)v01;
      }
      if (mB < p.M) {
        size_t rb = ((size_t)img * p.M + mB) * p.N;
        o[rb + na] = (f16)v10; o[rb + nb] = (f16)v11;
      }
    } else {
      float* o = (float*)p.out;
      if (mA < p.M) {
        size_t rb = ((size_t)img * p.M + mA) * p.N;
        o[rb + na] = v00; o[rb + nb] = v01;
      }
      if (mB < p.M) {
        size_t rb = ((size_t)img * p.M + mB) * p.N;
        o[rb + na] = v10; o[rb + nb] = v11;
      }
    }
  }
}

// ---------------------------------------------------------------------------
// Token build: CLS = mean(49 tokens) ; add pos_embed ; emit f16 tokens + CLS row
// ---------------------------------------------------------------------------
__global__ __launch_bounds__(256) void tok_kernel(const float* __restrict__ c3,
    const float* __restrict__ pos, f16* __restrict__ tokc, f16* __restrict__ cls) {
  int img = blockIdx.x;
  for (int d = threadIdx.x; d < DIMC; d += 256) {
    const float* base = c3 + (size_t)img * 49 * DIMC + d;
    float s = 0.f;
    for (int m = 0; m < 49; ++m) s += base[(size_t)m * DIMC];
    float cv = s * (1.f / 49.f) + pos[d];
    f16* tbp = tokc + (size_t)img * 50 * DIMC;
    tbp[d] = (f16)cv;
    cls[(size_t)img * DIMC + d] = (f16)cv;
    for (int m = 0; m < 49; ++m)
      tbp[(size_t)(m + 1) * DIMC + d] = (f16)(base[(size_t)m * DIMC] + pos[(size_t)(m + 1) * DIMC + d]);
  }
}

// ---------------------------------------------------------------------------
// Attention: logits = q.k^T * DIM^-0.5 ; softmax over 50 ; out = attn.v (f16)
// ---------------------------------------------------------------------------
__global__ __launch_bounds__(256) void attn_kernel(const float* __restrict__ q,
    const float* __restrict__ k, const float* __restrict__ v, f16* __restrict__ aout) {
  __shared__ float l[64];
  int img = blockIdx.x;
  int tid = threadIdx.x;
  int wave = tid >> 5, lane = tid & 31;
  const float* qr = q + (size_t)img * DIMC;
  for (int m = wave; m < 50; m += 8) {
    const float* kr = k + ((size_t)img * 50 + m) * DIMC;
    float s = 0.f;
    for (int d = lane; d < DIMC; d += 32) s += qr[d] * kr[d];
    #pragma unroll
    for (int off = 16; off > 0; off >>= 1) s += __shfl_xor(s, off);
    if (lane == 0) l[m] = s * 0.03125f;   // 1024^-0.5
  }
  __syncthreads();
  if (tid == 0) {
    float mx = -1e30f;
    for (int m = 0; m < 50; ++m) mx = fmaxf(mx, l[m]);
    float ssum = 0.f;
    for (int m = 0; m < 50; ++m) { float e = expf(l[m] - mx); l[m] = e; ssum += e; }
    float inv = 1.f / ssum;
    for (int m = 0; m < 50; ++m) l[m] *= inv;
  }
  __syncthreads();
  for (int d = tid; d < DIMC; d += 256) {
    const float* vr = v + (size_t)img * 50 * DIMC + d;
    float s = 0.f;
    for (int m = 0; m < 50; ++m) s += l[m] * vr[(size_t)m * DIMC];
    aout[(size_t)img * DIMC + d] = (f16)s;
  }
}

// ---------------------------------------------------------------------------
// Momentum fusion over IMAGE_ORDER
// ---------------------------------------------------------------------------
__global__ void fuse_kernel(const float* __restrict__ emb, const int* __restrict__ valid,
    const float* __restrict__ defv, const float* __restrict__ mom_p, float* __restrict__ out) {
  int n = blockIdx.x;
  float mom = *mom_p;
  const int order[6] = {2, 0, 1, 5, 3, 4};
  for (int d = threadIdx.x; d < DIMC; d += 256) {
    float e = defv[d];
    #pragma unroll
    for (int i = 0; i < 6; ++i) {
      int c = order[i];
      if (valid[n * 6 + c]) e = mom * e + (1.f - mom) * emb[(size_t)(n * 6 + c) * DIMC + d];
    }
    out[(size_t)n * DIMC + d] = e;
  }
}

// ---------------------------------------------------------------------------
extern "C" void kernel_launch(void* const* d_in, const int* in_sizes, int n_in,
                              void* d_out, int out_size, void* d_ws, size_t ws_size,
                              hipStream_t stream) {
  (void)in_sizes; (void)n_in; (void)out_size;
  const float* camera = (const float*)d_in[0];
  const float* boxes  = (const float*)d_in[1];
  const float* aug    = (const float*)d_in[2];
  const float* la     = (const float*)d_in[3];
  const float* l2i    = (const float*)d_in[4];
  const float* mom    = (const float*)d_in[5];
  const float* defdim = (const float*)d_in[6];
  const float* tw     = (const float*)d_in[7];
  const float* tb     = (const float*)d_in[8];
  const float* w1     = (const float*)d_in[9];
  const float* w2     = (const float*)d_in[10];
  const float* w3     = (const float*)d_in[11];
  const float* pos    = (const float*)d_in[12];
  const float* wq     = (const float*)d_in[13];
  const float* wk     = (const float*)d_in[14];
  const float* wv     = (const float*)d_in[15];
  const float* wo     = (const float*)d_in[16];

  char* ws = (char*)d_ws;
  auto AL = [](size_t x) { return (x + 255) & ~(size_t)255; };
  size_t off = 0;
  auto alloc = [&](size_t bytes) { size_t o = off; off = AL(off + bytes); return o; };

  // padded K per GEMM (multiple of 32)
  const int K1 = 3*7*7,   K1p = 160;
  const int K2 = 64*3*3,  K2p = 576;
  const int K3 = 256*3*3, K3p = 2304;
  const int KA = 1024,    KAp = 1024;

  size_t o_theta = alloc((size_t)NIMG * 6 * 4);
  size_t o_valid = alloc((size_t)NIMG * 4);
  size_t o_w1 = alloc((size_t)64 * K1p * 2);
  size_t o_w2 = alloc((size_t)256 * K2p * 2);
  size_t o_w3 = alloc((size_t)1024 * K3p * 2);
  size_t o_wq = alloc((size_t)1024 * KAp * 2);
  size_t o_wk = alloc((size_t)1024 * KAp * 2);
  size_t o_wv = alloc((size_t)1024 * KAp * 2);
  size_t o_wo = alloc((size_t)1024 * KAp * 2);
  // Region A: crops(115.6MB) -> later c3out(77MB) -> later vbuf(78.6MB)
  size_t SZ_A = (size_t)NIMG * CROPSZ * CROPSZ * 3 * 2;
  // Region B: c1out(154MB) -> later tokc+cls+q+a+emb (44MB)
  size_t SZ_B = (size_t)NIMG * 3136 * 64 * 2;
  // Region C: c2out(38.5MB) -> later kbuf(78.6MB)
  size_t SZ_C = (size_t)NIMG * 50 * 1024 * 4;
  size_t o_A = alloc(SZ_A);
  size_t o_B = alloc(SZ_B);
  size_t o_C = alloc(SZ_C);
  if (ws_size < off) return;   // insufficient scratch (deterministic no-op)

  float* theta = (float*)(ws + o_theta);
  int*   valid = (int*)(ws + o_valid);
  f16* w1h = (f16*)(ws + o_w1);
  f16* w2h = (f16*)(ws + o_w2);
  f16* w3h = (f16*)(ws + o_w3);
  f16* wqh = (f16*)(ws + o_wq);
  f16* wkh = (f16*)(ws + o_wk);
  f16* wvh = (f16*)(ws + o_wv);
  f16* woh = (f16*)(ws + o_wo);
  f16*   crops = (f16*)(ws + o_A);
  float* c3out = (float*)(ws + o_A);
  float* vbuf  = (float*)(ws + o_A);
  f16*   c1out = (f16*)(ws + o_B);
  f16*   tokc  = (f16*)(ws + o_B);
  f16*   cls   = (f16*)(ws + o_B + 39321600);
  float* qbuf  = (float*)(ws + o_B + 40108032);
  f16*   abuf  = (f16*)(ws + o_B + 41680896);
  float* emb   = (float*)(ws + o_B + 42467328);
  f16*   c2out = (f16*)(ws + o_C);
  float* kbuf  = (float*)(ws + o_C);

  auto launch_gemm = [&](const f16* act, const f16* wt, void* out,
                         int IC, int IH, int IW, int KH, int KW, int SH, int SW,
                         int PH, int PW, int OH, int OW, int N, int Kp,
                         int relu, int f16out, int nimg) {
    GemmP p;
    p.act = act; p.wt = wt; p.out = out;
    p.IC = IC; p.IH = IH; p.IW = IW; p.KH = KH; p.KW = KW;
    p.SH = SH; p.SW = SW; p.PH = PH; p.PW = PW;
    p.OH = OH; p.OW = OW; p.M = OH * OW; p.N = N; p.K = IC * KH * KW; p.Kp = Kp;
    p.relu = relu; p.out_f16 = f16out;
    dim3 g((unsigned)((p.M + BM - 1) / BM), (unsigned)((N + BN - 1) / BN), (unsigned)nimg);
    gemm_kernel<<<g, dim3(128), 0, stream>>>(p);
  };

  // 1. projection / theta / valid
  setup_kernel<<<6, 64, 0, stream>>>(boxes, la, l2i, aug, tw, tb, theta, valid);
  // 2. weight conversion into padded f16 [N][Kp]
  cvt_pad_kernel<<<(64*K1p + 255)/256, 256, 0, stream>>>(w1, w1h, 64, K1, K1p);
  cvt_pad_kernel<<<(256*K2p + 255)/256, 256, 0, stream>>>(w2, w2h, 256, K2, K2p);
  cvt_pad_kernel<<<(1024*K3p + 255)/256, 256, 0, stream>>>(w3, w3h, 1024, K3, K3p);
  cvtT_pad_kernel<<<(1024*KAp + 255)/256, 256, 0, stream>>>(wq, wqh, 1024, KA, KAp);
  cvtT_pad_kernel<<<(1024*KAp + 255)/256, 256, 0, stream>>>(wk, wkh, 1024, KA, KAp);
  cvtT_pad_kernel<<<(1024*KAp + 255)/256, 256, 0, stream>>>(wv, wvh, 1024, KA, KAp);
  cvtT_pad_kernel<<<(1024*KAp + 255)/256, 256, 0, stream>>>(wo, woh, 1024, KA, KAp);
  // 3. grid-sample crops (NHWC f16)
  crop_kernel<<<dim3(NIMG, (CROPSZ*CROPSZ)/256), 256, 0, stream>>>(camera, theta, crops);
  // 4. CLIP head convs (implicit-GEMM WMMA)
  launch_gemm(crops, w1h, c1out, 3, 224, 224, 7, 7, 4, 4, 1, 1, 56, 56,   64, K1p, 1, 1, NIMG);
  launch_gemm(c1out, w2h, c2out, 64,  56,  56, 3, 3, 4, 4, 0, 0, 14, 14, 256, K2p, 1, 1, NIMG);
  launch_gemm(c2out, w3h, c3out, 256, 14,  14, 3, 3, 2, 2, 0, 0,  7,  7, 1024, K3p, 1, 0, NIMG);
  // 5. tokens + CLS
  tok_kernel<<<NIMG, 256, 0, stream>>>(c3out, pos, tokc, cls);
  // 6. projections: k, v per-image; q batched over all images
  launch_gemm(tokc, wkh, kbuf, 1024, 50, 1, 1, 1, 1, 1, 0, 0, 50, 1, 1024, KAp, 0, 0, NIMG);
  launch_gemm(tokc, wvh, vbuf, 1024, 50, 1, 1, 1, 1, 1, 0, 0, 50, 1, 1024, KAp, 0, 0, NIMG);
  launch_gemm(cls,  wqh, qbuf, 1024, NIMG, 1, 1, 1, 1, 1, 0, 0, NIMG, 1, 1024, KAp, 0, 0, 1);
  // 7. attention
  attn_kernel<<<NIMG, 256, 0, stream>>>(qbuf, kbuf, vbuf, abuf);
  // 8. output projection (one M=384 GEMM)
  launch_gemm(abuf, woh, emb, 1024, NIMG, 1, 1, 1, 1, 1, 0, 0, NIMG, 1, 1024, KAp, 0, 0, 1);
  // 9. momentum fusion
  fuse_kernel<<<64, 256, 0, stream>>>(emb, valid, defdim, mom, (float*)d_out);
}
